// CVDotProductAttention_81544249081935
// MI455X (gfx1250) — compile-verified, matrix operand order checked by reading, not run on hardware
//
#include <hip/hip_runtime.h>

#define BB 4
#define SS 4096
#define DD 64
#define KCHUNK 64
#define WAVES 4
#define QTILE (WAVES * 16)

typedef __bf16 bf16;
typedef __bf16 v2bf  __attribute__((ext_vector_type(2)));
typedef __bf16 v16bf __attribute__((ext_vector_type(16)));
typedef float v8f __attribute__((ext_vector_type(8)));

__device__ __forceinline__ v8f wmma_bf16(v16bf a, v16bf b, v8f c) {
    // D = A(16x32 bf16) * B(32x16 bf16) + C(16x16 f32)
    return __builtin_amdgcn_wmma_f32_16x16x32_bf16(false, a, false, b, (short)0, c, false, false);
}

// Single-instruction hardware transcendentals (v_sqrt_f32 / v_exp_f32 / v_rcp_f32).
__device__ __forceinline__ float hw_sqrt(float x) { return __builtin_amdgcn_sqrtf(x); }
__device__ __forceinline__ float hw_exp2(float x) { return __builtin_amdgcn_exp2f(x); }
__device__ __forceinline__ float hw_rcp(float x)  { return __builtin_amdgcn_rcpf(x); }

// Packed fp32x2 -> bf16x2 conversion (v_cvt_pk_bf16_f32 when available).
__device__ __forceinline__ v2bf pk_bf16(float lo, float hi) {
#if __has_builtin(__builtin_amdgcn_cvt_pk_bf16_f32)
    return __builtin_amdgcn_cvt_pk_bf16_f32(lo, hi);
#else
    v2bf r;
    r.x = (bf16)lo;
    r.y = (bf16)hi;
    return r;
#endif
}

// Build a 16-bf16 fragment from two contiguous 8-element (16B) LDS regions.
__device__ __forceinline__ v16bf lds_frag(const bf16* p0, const bf16* p1) {
    union { v16bf v; uint4 q[2]; } r;
    r.q[0] = *(const uint4*)p0;
    r.q[1] = *(const uint4*)p1;
    return r.v;
}

// Build a fragment from a global fp32 row, converting to bf16.
__device__ __forceinline__ v16bf f32_frag(const float* row, int off0, int off1) {
    v16bf f;
#pragma unroll
    for (int j = 0; j < 8; ++j) f[j] = (bf16)row[off0 + j];
#pragma unroll
    for (int j = 0; j < 8; ++j) f[8 + j] = (bf16)row[off1 + j];
    return f;
}

// Negate all 16 packed bf16 values (sign-bit flip on the packed dwords).
__device__ __forceinline__ v16bf neg_frag(v16bf a) {
    union { v16bf v; unsigned u[8]; } x;
    x.v = a;
#pragma unroll
    for (int j = 0; j < 8; ++j) x.u[j] ^= 0x80008000u;
    return x.v;
}

__global__ __launch_bounds__(WAVES * 32, 1) void cv_attn_kernel(
    const float* __restrict__ q_re, const float* __restrict__ q_im,
    const float* __restrict__ k_re, const float* __restrict__ k_im,
    const float* __restrict__ v_re, const float* __restrict__ v_im,
    float* __restrict__ out)
{
    // K chunk (row-major [key][d]) and V chunk transposed ([d][key]), bf16.
    __shared__ __align__(16) bf16 sKre[KCHUNK * DD];
    __shared__ __align__(16) bf16 sKim[KCHUNK * DD];
    __shared__ __align__(16) bf16 sVtr[DD * KCHUNK];
    __shared__ __align__(16) bf16 sVti[DD * KCHUNK];
    __shared__ __align__(16) bf16 sP[WAVES][16 * KCHUNK]; // per-wave P transpose scratch

    const int tid  = threadIdx.x;
    const int wave = tid >> 5;
    const int lane = tid & 31;
    const int nlo  = lane & 15;        // column / M-row selector within a 16-lane half
    const int hi   = (lane >> 4) & 1;  // which lane-half
    const int koff = hi * 8;           // K-offset split per the 16-bit fragment layout

    const int nQblk = SS / QTILE;
    const int b  = blockIdx.x / nQblk;
    const int qb = blockIdx.x % nQblk;
    const int q0 = qb * QTILE + wave * 16;

    // softmax in exp2 domain: logits pre-scaled by 1/sqrt(D) * log2(e)
    const float scaleL2 = 0.125f * 1.4426950408889634f;

    // ---- persistent Q fragments (A-matrix layout: lane%16 = M row) ----
    const float* qr_row = q_re + ((size_t)b * SS + q0 + nlo) * DD;
    const float* qi_row = q_im + ((size_t)b * SS + q0 + nlo) * DD;
    v16bf aQr0 = f32_frag(qr_row, koff,      16 + koff); // d 0..31
    v16bf aQr1 = f32_frag(qr_row, 32 + koff, 48 + koff); // d 32..63
    v16bf aQi0 = f32_frag(qi_row, koff,      16 + koff);
    v16bf aQi1 = f32_frag(qi_row, 32 + koff, 48 + koff);
    v16bf aQn0 = neg_frag(aQi0);
    v16bf aQn1 = neg_frag(aQi1);

    const v8f zero = {0.f, 0.f, 0.f, 0.f, 0.f, 0.f, 0.f, 0.f};
    v8f o_re[4], o_im[4];
#pragma unroll
    for (int t = 0; t < 4; ++t) { o_re[t] = zero; o_im[t] = zero; }

    float m_s[8], l_s[8];
#pragma unroll
    for (int v = 0; v < 8; ++v) { m_s[v] = -3.0e38f; l_s[v] = 0.f; }

    const float* gKr = k_re + (size_t)b * SS * DD;
    const float* gKi = k_im + (size_t)b * SS * DD;
    const float* gVr = v_re + (size_t)b * SS * DD;
    const float* gVi = v_im + (size_t)b * SS * DD;

#pragma unroll 1
    for (int kc = 0; kc < SS; kc += KCHUNK) {
        __syncthreads(); // previous chunk fully consumed before overwrite

        // ---- cooperative stage of K/V chunk into LDS (fp32 -> bf16) ----
        const float* pkr = gKr + (size_t)kc * DD;
        const float* pki = gKi + (size_t)kc * DD;
        const float* pvr = gVr + (size_t)kc * DD;
        const float* pvi = gVi + (size_t)kc * DD;
#pragma unroll 2
        for (int i = 0; i < (KCHUNK * DD) / (WAVES * 32 * 4); ++i) {
            const int e   = (i * WAVES * 32 + tid) * 4;
            const int key = e >> 6;
            const int d   = e & 63;
            float4 fr = *(const float4*)(pkr + e);
            float4 fi = *(const float4*)(pki + e);
            // packed converts + 2-dword LDS stores for the row-major K tiles
            v2bf* dr = (v2bf*)&sKre[key * DD + d];
            dr[0] = pk_bf16(fr.x, fr.y);
            dr[1] = pk_bf16(fr.z, fr.w);
            v2bf* di = (v2bf*)&sKim[key * DD + d];
            di[0] = pk_bf16(fi.x, fi.y);
            di[1] = pk_bf16(fi.z, fi.w);
            float4 vr4 = *(const float4*)(pvr + e);
            float4 vi4 = *(const float4*)(pvi + e);
            sVtr[(d + 0) * KCHUNK + key] = (bf16)vr4.x;
            sVtr[(d + 1) * KCHUNK + key] = (bf16)vr4.y;
            sVtr[(d + 2) * KCHUNK + key] = (bf16)vr4.z;
            sVtr[(d + 3) * KCHUNK + key] = (bf16)vr4.w;
            sVti[(d + 0) * KCHUNK + key] = (bf16)vi4.x;
            sVti[(d + 1) * KCHUNK + key] = (bf16)vi4.y;
            sVti[(d + 2) * KCHUNK + key] = (bf16)vi4.z;
            sVti[(d + 3) * KCHUNK + key] = (bf16)vi4.w;
        }
        __syncthreads();

        // prefetch next chunk while we compute
        if (kc + KCHUNK < SS) {
            const size_t nx = (size_t)(kc + KCHUNK) * DD + (size_t)tid * 32;
            __builtin_prefetch(gKr + nx, 0, 1);
            __builtin_prefetch(gKi + nx, 0, 1);
            __builtin_prefetch(gVr + nx, 0, 1);
            __builtin_prefetch(gVi + nx, 0, 1);
        }

        // ---- complex scores -> log2-domain magnitudes, 4 subtiles of 16 keys ----
        float mag[4][8];
#pragma unroll
        for (int nt = 0; nt < 4; ++nt) {
            const bf16* krow = &sKre[(nt * 16 + nlo) * DD]; // B frag: lane = key column
            const bf16* irow = &sKim[(nt * 16 + nlo) * DD];
            v16bf bR0 = lds_frag(krow + koff,      krow + 16 + koff);
            v16bf bR1 = lds_frag(krow + 32 + koff, krow + 48 + koff);
            v16bf bI0 = lds_frag(irow + koff,      irow + 16 + koff);
            v16bf bI1 = lds_frag(irow + 32 + koff, irow + 48 + koff);
            v8f sre = zero, sim = zero;
            sre = wmma_bf16(aQr0, bR0, sre);   // qr*kr (d lo)
            sre = wmma_bf16(aQr1, bR1, sre);   // qr*kr (d hi)
            sre = wmma_bf16(aQn0, bI0, sre);   // -qi*ki
            sre = wmma_bf16(aQn1, bI1, sre);
            sim = wmma_bf16(aQr0, bI0, sim);   // qr*ki
            sim = wmma_bf16(aQr1, bI1, sim);
            sim = wmma_bf16(aQi0, bR0, sim);   // qi*kr
            sim = wmma_bf16(aQi1, bR1, sim);
            // scalar fma form: avoids pk-pairing register shuffles on the
            // two disjoint accumulator ranges
#pragma unroll
            for (int v = 0; v < 8; ++v) {
                float re = sre[v], im = sim[v];
                float m2 = __builtin_fmaf(re, re, im * im);
                mag[nt][v] = scaleL2 * hw_sqrt(m2);
            }
        }

        // ---- online softmax over |s| (base-2) ----
        float mnew[8];
#pragma unroll
        for (int v = 0; v < 8; ++v)
            mnew[v] = fmaxf(fmaxf(fmaxf(mag[0][v], mag[1][v]),
                                  fmaxf(mag[2][v], mag[3][v])), m_s[v]);
#pragma unroll
        for (int msk = 1; msk <= 8; msk <<= 1)
#pragma unroll
            for (int v = 0; v < 8; ++v)
                mnew[v] = fmaxf(mnew[v], __shfl_xor(mnew[v], msk, 32));

        float alpha[8], psum[8];
#pragma unroll
        for (int v = 0; v < 8; ++v) {
            alpha[v] = hw_exp2(m_s[v] - mnew[v]);
            m_s[v]   = mnew[v];
            psum[v]  = 0.f;
        }

        bf16* pP = sP[wave];
#pragma unroll
        for (int nt = 0; nt < 4; ++nt)
#pragma unroll
            for (int v = 0; v < 8; ++v) {
                float p = hw_exp2(mag[nt][v] - m_s[v]);
                psum[v] += p;
                // C layout (lane=N, vgpr=M) -> row-major [m][k] in LDS
                pP[(v + hi * 8) * KCHUNK + nt * 16 + nlo] = (bf16)p;
            }

#pragma unroll
        for (int msk = 1; msk <= 8; msk <<= 1)
#pragma unroll
            for (int v = 0; v < 8; ++v)
                psum[v] += __shfl_xor(psum[v], msk, 32);

#pragma unroll
        for (int v = 0; v < 8; ++v) l_s[v] = l_s[v] * alpha[v] + psum[v];
#pragma unroll
        for (int t = 0; t < 4; ++t)
#pragma unroll
            for (int v = 0; v < 8; ++v) {
                o_re[t][v] *= alpha[v];
                o_im[t][v] *= alpha[v];
            }

        // ---- PV: O += P(16x64) @ V(64xD), via two K=32 halves ----
        const bf16* prow = pP + nlo * KCHUNK; // A frag: lane = M row
        v16bf pA0 = lds_frag(prow + koff,      prow + 16 + koff); // keys 0..31
        v16bf pA1 = lds_frag(prow + 32 + koff, prow + 48 + koff); // keys 32..63
#pragma unroll
        for (int t = 0; t < 4; ++t) {
            const bf16* vrr = &sVtr[(t * 16 + nlo) * KCHUNK]; // B frag: lane = d column
            const bf16* vri = &sVti[(t * 16 + nlo) * KCHUNK];
            v16bf bVr0 = lds_frag(vrr + koff,      vrr + 16 + koff);
            v16bf bVr1 = lds_frag(vrr + 32 + koff, vrr + 48 + koff);
            v16bf bVi0 = lds_frag(vri + koff,      vri + 16 + koff);
            v16bf bVi1 = lds_frag(vri + 32 + koff, vri + 48 + koff);
            o_re[t] = wmma_bf16(pA0, bVr0, o_re[t]);
            o_re[t] = wmma_bf16(pA1, bVr1, o_re[t]);
            o_im[t] = wmma_bf16(pA0, bVi0, o_im[t]);
            o_im[t] = wmma_bf16(pA1, bVi1, o_im[t]);
        }
    }

    // ---- epilogue: normalize and write [2,B,S,D] ----
    float invl[8];
#pragma unroll
    for (int v = 0; v < 8; ++v) invl[v] = hw_rcp(l_s[v]);

    float* out_re = out;
    float* out_im = out + (size_t)BB * SS * DD;
#pragma unroll
    for (int t = 0; t < 4; ++t)
#pragma unroll
        for (int v = 0; v < 8; ++v) {
            const int q = q0 + v + hi * 8;
            const int d = t * 16 + nlo;
            const size_t off = ((size_t)b * SS + q) * DD + d;
            out_re[off] = o_re[t][v] * invl[v];
            out_im[off] = o_im[t][v] * invl[v];
        }
}

extern "C" void kernel_launch(void* const* d_in, const int* in_sizes, int n_in,
                              void* d_out, int out_size, void* d_ws, size_t ws_size,
                              hipStream_t stream) {
    const float* q_re = (const float*)d_in[0];
    const float* q_im = (const float*)d_in[1];
    const float* k_re = (const float*)d_in[2];
    const float* k_im = (const float*)d_in[3];
    const float* v_re = (const float*)d_in[4];
    const float* v_im = (const float*)d_in[5];
    (void)in_sizes; (void)n_in; (void)out_size; (void)d_ws; (void)ws_size;

    dim3 grid(BB * (SS / QTILE)); // 4 * 64 = 256 blocks
    dim3 block(WAVES * 32);       // 4 wave32s
    hipLaunchKernelGGL(cv_attn_kernel, grid, block, 0, stream,
                       q_re, q_im, k_re, k_im, v_re, v_im, (float*)d_out);
}
